// scrim_41540923686996
// MI455X (gfx1250) — compile-verified, matrix operand order checked by reading, not run on hardware
//
#include <hip/hip_runtime.h>
#include <hip/hip_bf16.h>

// ---------------------------------------------------------------------------
// MI455X (gfx1250) implementation.
//
// h = relu(x_cat[100,20064] @ fc1_w.T) streams ~10.6 MB from HBM -> memory/
// latency bound (~0.5us floor at 23.3 TB/s). bf16 WMMA (v_wmma_f32_16x16x32
// _bf16) with split-K: 7x2x25 single-wave workgroups over the x range
// (K=20000), branch-free inner loop (padded rows clamped to row 99, embed
// tail handled by a separate 2-step WMMA kernel), f32 atomic reduction into
// scratch. M = h @ T is one more WMMA step. Tail MLP: one wave per row.
// ---------------------------------------------------------------------------

typedef __attribute__((ext_vector_type(16))) __bf16 v16bf;
typedef __attribute__((ext_vector_type(8)))  float  v8f;

namespace cfg {
constexpr int B     = 16384;   // batch rows in x
constexpr int D     = 20000;   // x feature dim = 625 * 32
constexpr int E     = 64;      // embed dim = 2 * 32
constexpr int KTOT  = D + E;   // 20064
constexpr int H0    = 32;
constexpr int H1    = 16;
constexpr int H2    = 8;
constexpr int OF    = 16;
constexpr int KD    = 3;
constexpr int NSUB  = 100;
constexpr int MPAD  = 112;     // 7 * 16
constexpr int MT    = 7;       // M tiles of 16
constexpr int XSTEPS = D / 32;      // 625
constexpr int KCHUNK = 25;          // steps per split-K chunk
constexpr int NCHUNK = 25;          // 25 * 25 = 625
}

// Deterministic stand-in for randperm(16384)[:100]: bijective affine map mod 2^14.
__device__ __forceinline__ int sub_idx(int r) {
    return (r * 40503 + 2077) & (cfg::B - 1);
}

// Load 16 bf16 fragment elements for one lane per the ISA 16-bit 16x32 layout:
// e=0..7 from K = khalf+e, e=8..15 from K = 16+khalf+(e-8); two contiguous
// 8-float runs -> four 128-bit loads.
__device__ __forceinline__ v16bf frag_from_f32(const float* __restrict__ p, int khalf) {
    const float4* q0 = reinterpret_cast<const float4*>(p + khalf);
    const float4* q1 = reinterpret_cast<const float4*>(p + 16 + khalf);
    float4 f0 = q0[0], f1 = q0[1];
    float4 f2 = q1[0], f3 = q1[1];
    v16bf v;
    v[0]  = (__bf16)f0.x; v[1]  = (__bf16)f0.y; v[2]  = (__bf16)f0.z; v[3]  = (__bf16)f0.w;
    v[4]  = (__bf16)f1.x; v[5]  = (__bf16)f1.y; v[6]  = (__bf16)f1.z; v[7]  = (__bf16)f1.w;
    v[8]  = (__bf16)f2.x; v[9]  = (__bf16)f2.y; v[10] = (__bf16)f2.z; v[11] = (__bf16)f2.w;
    v[12] = (__bf16)f3.x; v[13] = (__bf16)f3.y; v[14] = (__bf16)f3.z; v[15] = (__bf16)f3.w;
    return v;
}

// --------------------------- kernel 0: zero scratch ------------------------
__global__ __launch_bounds__(256)
void k_zero(float* __restrict__ acc, int n) {
    for (int i = blockIdx.x * blockDim.x + threadIdx.x; i < n; i += gridDim.x * blockDim.x)
        acc[i] = 0.0f;
}

// ------------- kernel 1a: split-K bf16 WMMA GEMM over x range --------------
// grid = (7 Mtiles, 2 Ntiles, 25 Kchunks), block = 32 (one wave).
// Branch-free inner loop: padded rows (>=100) are clamped to row 99; their
// accumulator rows are garbage and never read downstream.
__global__ __launch_bounds__(32)
void k_gemm1_x(const float* __restrict__ x,
               const float* __restrict__ fc1_w,
               float*       __restrict__ acc_out)   // [MPAD][H0] f32
{
    const int lane  = threadIdx.x;
    const int mt    = blockIdx.x;
    const int nt    = blockIdx.y;
    const int kc    = blockIdx.z;
    const int lrow  = lane & 15;
    const int khalf = (lane >> 4) << 3;          // 0 or 8

    const int r  = mt * 16 + lrow;
    const int rc = (r < cfg::NSUB) ? r : (cfg::NSUB - 1);   // clamp padding
    const float* xrow = x + (size_t)sub_idx(rc) * cfg::D;
    const float* wrow = fc1_w + (size_t)(nt * 16 + lrow) * cfg::KTOT;

    v8f acc = {};
    const int s0 = kc * cfg::KCHUNK;
#pragma unroll 1
    for (int s = s0; s < s0 + cfg::KCHUNK; ++s) {
        const int k0 = s * 32;
        __builtin_prefetch(wrow + k0 + 32, 0, 1);   // global_prefetch_b8
        __builtin_prefetch(xrow + k0 + 32, 0, 1);
        v16bf afr = frag_from_f32(xrow + k0, khalf);
        v16bf bfr = frag_from_f32(wrow + k0, khalf);
        acc = __builtin_amdgcn_wmma_f32_16x16x32_bf16(
            false, afr, false, bfr, (short)0, acc, false, false);
    }

    // D layout: VGPR v, lanes 0-15 -> M=v, lanes 16-31 -> M=8+v ; N = lane&15.
    const int colg  = nt * 16 + lrow;
    const int rbase = mt * 16 + ((lane >> 4) << 3);
#pragma unroll
    for (int v = 0; v < 8; ++v)
        atomicAdd(acc_out + (size_t)(rbase + v) * cfg::H0 + colg, acc[v]);
}

// ------------- kernel 1b: embed tail (K = 20000..20063), 2 steps -----------
// grid = (7 Mtiles, 2 Ntiles), block = 32.
__global__ __launch_bounds__(32)
void k_gemm1_e(const int*   __restrict__ labels,
               const float* __restrict__ embed,
               const float* __restrict__ fc1_w,
               float*       __restrict__ acc_out)
{
    const int lane  = threadIdx.x;
    const int mt    = blockIdx.x;
    const int nt    = blockIdx.y;
    const int lrow  = lane & 15;
    const int khalf = (lane >> 4) << 3;

    const int r  = mt * 16 + lrow;
    const int rc = (r < cfg::NSUB) ? r : (cfg::NSUB - 1);
    const int gr = sub_idx(rc);
    const float* erow = embed + (size_t)labels[gr] * cfg::E;
    const float* wrow = fc1_w + (size_t)(nt * 16 + lrow) * cfg::KTOT + cfg::D;

    v8f acc = {};
#pragma unroll
    for (int s = 0; s < 2; ++s) {
        const int k0 = s * 32;
        v16bf afr = frag_from_f32(erow + k0, khalf);
        v16bf bfr = frag_from_f32(wrow + k0, khalf);
        acc = __builtin_amdgcn_wmma_f32_16x16x32_bf16(
            false, afr, false, bfr, (short)0, acc, false, false);
    }

    const int colg  = nt * 16 + lrow;
    const int rbase = mt * 16 + ((lane >> 4) << 3);
#pragma unroll
    for (int v = 0; v < 8; ++v)
        atomicAdd(acc_out + (size_t)(rbase + v) * cfg::H0 + colg, acc[v]);
}

// ----------------- kernel 2: bias + relu into h [NSUB][H0] -----------------
__global__ __launch_bounds__(256)
void k_finalize_h(const float* __restrict__ acc,
                  const float* __restrict__ fc1_b,
                  float* __restrict__ h)
{
    int i = blockIdx.x * blockDim.x + threadIdx.x;
    if (i < cfg::NSUB * cfg::H0) {
        int col = i & (cfg::H0 - 1);
        h[i] = fmaxf(acc[i] + fc1_b[col], 0.0f);
    }
}

// -------- kernel 3: M = h[100,32] @ T.reshape(32,48) via one WMMA step -----
// grid = (7 Mtiles, 3 Ntiles), block = 32.
__global__ __launch_bounds__(32)
void k_gemm_M(const float* __restrict__ h,
              const float* __restrict__ Tm,     // [32][48] row-major
              float* __restrict__ Mout)         // [NSUB][48]
{
    const int lane  = threadIdx.x;
    const int mt    = blockIdx.x;
    const int nt    = blockIdx.y;
    const int lrow  = lane & 15;
    const int khalf = (lane >> 4) << 3;

    const int r  = mt * 16 + lrow;
    const int rc = (r < cfg::NSUB) ? r : (cfg::NSUB - 1);
    v16bf afr = frag_from_f32(h + (size_t)rc * cfg::H0, khalf);

    const int n = nt * 16 + lrow;               // output column 0..47
    v16bf bfr;
#pragma unroll
    for (int e = 0; e < 16; ++e) {
        int kl = (e < 8) ? (khalf + e) : (8 + khalf + e);   // 16+khalf+(e-8)
        bfr[e] = (__bf16)Tm[kl * (cfg::OF * cfg::KD) + n];
    }

    v8f acc = {};
    acc = __builtin_amdgcn_wmma_f32_16x16x32_bf16(
        false, afr, false, bfr, (short)0, acc, false, false);

    const int colg  = nt * 16 + lrow;
    const int rbase = mt * 16 + ((lane >> 4) << 3);
#pragma unroll
    for (int v = 0; v < 8; ++v) {
        int row = rbase + v;
        if (row < cfg::NSUB)
            Mout[(size_t)row * (cfg::OF * cfg::KD) + colg] = acc[v];
    }
}

// ------ kernel 4: minibatch-disc + fc2/fc3/fc4, one wave per output row ----
__global__ __launch_bounds__(32)
void k_mbd_mlp(const float* __restrict__ Mfeat,  // [NSUB][48]
               const float* __restrict__ h,      // [NSUB][32]
               const float* __restrict__ fc2_w, const float* __restrict__ fc2_b,
               const float* __restrict__ fc3_w, const float* __restrict__ fc3_b,
               const float* __restrict__ fc4_w, const float* __restrict__ fc4_b,
               float* __restrict__ out)          // [NSUB]
{
    __shared__ float hc[cfg::H0 + cfg::OF];      // 48
    __shared__ float h2[cfg::H1];
    __shared__ float h3[cfg::H2];

    const int i = blockIdx.x;
    const int t = threadIdx.x;

    hc[t] = h[(size_t)i * cfg::H0 + t];          // t in [0,32)

    if (t < cfg::OF) {
        const float* mi = Mfeat + (size_t)i * 48 + t * cfg::KD;
        float m0 = mi[0], m1 = mi[1], m2 = mi[2];
        float s = 0.0f;
        for (int j = 0; j < cfg::NSUB; ++j) {
            const float* mj = Mfeat + (size_t)j * 48 + t * cfg::KD;
            float d = fabsf(m0 - mj[0]) + fabsf(m1 - mj[1]) + fabsf(m2 - mj[2]);
            s += __expf(-d);
        }
        hc[cfg::H0 + t] = s;
    }
    __syncthreads();

    if (t < cfg::H1) {
        float s = fc2_b[t];
#pragma unroll
        for (int k = 0; k < 48; ++k) s += hc[k] * fc2_w[t * 48 + k];
        h2[t] = fmaxf(s, 0.0f);
    }
    __syncthreads();

    if (t < cfg::H2) {
        float s = fc3_b[t];
#pragma unroll
        for (int k = 0; k < cfg::H1; ++k) s += h2[k] * fc3_w[t * cfg::H1 + k];
        h3[t] = fmaxf(s, 0.0f);
    }
    __syncthreads();

    if (t == 0) {
        float s = fc4_b[0];
#pragma unroll
        for (int k = 0; k < cfg::H2; ++k) s += h3[k] * fc4_w[k];
        out[i] = s;
    }
}

// ---------------------------------------------------------------------------
extern "C" void kernel_launch(void* const* d_in, const int* in_sizes, int n_in,
                              void* d_out, int out_size, void* d_ws, size_t ws_size,
                              hipStream_t stream) {
    const float* x      = (const float*)d_in[0];
    const int*   labels = (const int*)  d_in[1];
    const float* embed  = (const float*)d_in[2];
    const float* fc1_w  = (const float*)d_in[3];
    const float* fc1_b  = (const float*)d_in[4];
    const float* Tm     = (const float*)d_in[5];
    const float* fc2_w  = (const float*)d_in[6];
    const float* fc2_b  = (const float*)d_in[7];
    const float* fc3_w  = (const float*)d_in[8];
    const float* fc3_b  = (const float*)d_in[9];
    const float* fc4_w  = (const float*)d_in[10];
    const float* fc4_b  = (const float*)d_in[11];
    float* out = (float*)d_out;

    // scratch layout
    float* ws_acc = (float*)d_ws;                              // MPAD*H0
    float* ws_h   = ws_acc + cfg::MPAD * cfg::H0;              // NSUB*H0
    float* ws_M   = ws_h   + cfg::NSUB * cfg::H0;              // NSUB*48

    const int accN = cfg::MPAD * cfg::H0;
    k_zero<<<dim3(4), dim3(256), 0, stream>>>(ws_acc, accN);

    k_gemm1_x<<<dim3(cfg::MT, cfg::H0 / 16, cfg::NCHUNK), dim3(32), 0, stream>>>(
        x, fc1_w, ws_acc);
    k_gemm1_e<<<dim3(cfg::MT, cfg::H0 / 16), dim3(32), 0, stream>>>(
        labels, embed, fc1_w, ws_acc);

    k_finalize_h<<<dim3((cfg::NSUB * cfg::H0 + 255) / 256), dim3(256), 0, stream>>>(
        ws_acc, fc1_b, ws_h);

    k_gemm_M<<<dim3(cfg::MT, 3), dim3(32), 0, stream>>>(ws_h, Tm, ws_M);

    k_mbd_mlp<<<dim3(cfg::NSUB), dim3(32), 0, stream>>>(
        ws_M, ws_h, fc2_w, fc2_b, fc3_w, fc3_b, fc4_w, fc4_b, out);
}
